// MockJEPADFoT_48404281425985
// MI455X (gfx1250) — matrix operand order repaired and checked by reading, not compiled, not run on hardware
//
#include <hip/hip_runtime.h>

// ---------------------------------------------------------------------------
// MI455X (gfx1250) wave32 WMMA implementation of the MockJEPA forward pass.
// All matmuls (convs as implicit GEMM, linears, attention) run through
// v_wmma_f32_16x16x32_f16 with f16 operands / f32 accumulation.
// Round 3: EDGE templating (bounds-free hot kernels) + ping-pong LDS double
// buffering (1 barrier per K-step, load/WMMA overlap).
// ---------------------------------------------------------------------------

typedef __attribute__((ext_vector_type(16))) _Float16 v16h;
typedef __attribute__((ext_vector_type(8)))  _Float16 v8h;
typedef __attribute__((ext_vector_type(4)))  _Float16 v4h;
typedef __attribute__((ext_vector_type(8)))  float    v8f;

__device__ __forceinline__ v8f wmma_f16f32(v16h a, v16h b, v8f c) {
  // 8 args: (neg_a, A, neg_b, B, c_mod, C, reuse_a, reuse_b)
  return __builtin_amdgcn_wmma_f32_16x16x32_f16(false, a, false, b, (short)0, c,
                                                false, false);
}

__device__ __forceinline__ v8h v8h_zero() {
  v8h z = {(_Float16)0, (_Float16)0, (_Float16)0, (_Float16)0,
           (_Float16)0, (_Float16)0, (_Float16)0, (_Float16)0};
  return z;
}

__device__ __forceinline__ v16h cat8(v8h lo, v8h hi) {
  return __builtin_shufflevector(lo, hi, 0, 1, 2, 3, 4, 5, 6, 7,
                                 8, 9, 10, 11, 12, 13, 14, 15);
}

__device__ __forceinline__ float act_apply(float x, int act) {
  if (act == 1) return fmaxf(x, 0.0f);                                    // relu
  if (act == 2) return 0.5f * x * (1.0f + erff(x * 0.7071067811865475f)); // gelu
  return x;
}

// ---------------------------------------------------------------------------
// Weight convert: plain f32 -> f16 copy (row-major N x K preserved; this IS
// the WMMA B-fragment layout: contiguous in K per output channel).
// ---------------------------------------------------------------------------
__global__ void jepa_w_f16(const float* __restrict__ W, _Float16* __restrict__ Wh,
                           int total) {
  int i4 = (blockIdx.x * blockDim.x + threadIdx.x) * 4;
  if (i4 + 3 < total) {
    float4 v = *(const float4*)(W + i4);
    v4h o;
    o[0] = (_Float16)v.x; o[1] = (_Float16)v.y;
    o[2] = (_Float16)v.z; o[3] = (_Float16)v.w;
    *(v4h*)(Wh + i4) = o;
  } else {
    for (int j = i4; j < total; ++j) Wh[j] = (_Float16)W[j];
  }
}

// ---------------------------------------------------------------------------
// Fragment helpers. B-fragment as used here: col = lane&15, k = (lane>>4)*16 + j
// (16 contiguous halfs -> two b128 loads from row-major N x K weights).
// A-fragment: row = lane&15, k = (lane>>4)*16 + j (matches B-fragment)
// (two contiguous 16B runs per lane from the LDS tile).
// ---------------------------------------------------------------------------
__device__ __forceinline__ v16h load_bfrag_fast(const _Float16* __restrict__ W,
                                                int n, int K, int kb0) {
  const _Float16* bp = W + (size_t)n * K + kb0;
  return cat8(*(const v8h*)bp, *(const v8h*)(bp + 8));
}

__device__ __forceinline__ v16h load_bfrag_edge(const _Float16* __restrict__ W,
                                                int n, int N, int K, int kb0) {
  v8h blo = v8h_zero(), bhi = v8h_zero();
  if (n < N) {
    const _Float16* bp = W + (size_t)n * K;
    if (kb0 + 16 <= K) {
      blo = *(const v8h*)(bp + kb0);
      bhi = *(const v8h*)(bp + kb0 + 8);
    } else {
      #pragma unroll
      for (int j = 0; j < 8; ++j) {
        int gk = kb0 + j;
        _Float16 v = bp[gk < K ? gk : 0];
        blo[j] = (gk < K) ? v : (_Float16)0;
      }
      #pragma unroll
      for (int j = 0; j < 8; ++j) {
        int gk = kb0 + 8 + j;
        _Float16 v = bp[gk < K ? gk : 0];
        bhi[j] = (gk < K) ? v : (_Float16)0;
      }
    }
  }
  return cat8(blo, bhi);
}

__device__ __forceinline__ v16h load_afrag(const _Float16* As, int mrow, int lhalf) {
  v8h lo = *(const v8h*)(As + mrow * 32 + lhalf * 16);
  v8h hi = *(const v8h*)(As + mrow * 32 + lhalf * 16 + 8);
  return cat8(lo, hi);
}

// ---------------------------------------------------------------------------
// Generic GEMM: C[M,N] = act(A[M,K] @ W[N,K]^T + bias + res)
// 256 threads = 8 waves; tile 64(M) x 32(N); K-step 32; A staged in LDS with
// ping-pong double buffering (EDGE=false), B fed straight from global.
// EDGE=false requires M%64==0, N%32==0, K%32==0, lda%8==0.
// ---------------------------------------------------------------------------
template <bool EDGE>
__global__ void __launch_bounds__(256)
jepa_gemm_f16(const _Float16* __restrict__ A, int lda,
              const _Float16* __restrict__ W,
              const float* __restrict__ bias,
              const float* __restrict__ res, int ldr,
              float* __restrict__ Cf, _Float16* __restrict__ Ch, int ldc,
              int M, int N, int K, int act) {
  __shared__ __align__(16) _Float16 As[2][64 * 32];

  const int tid   = threadIdx.x;
  const int lane  = tid & 31;
  const int wid   = tid >> 5;   // 0..7
  const int wm    = wid & 3;    // M sub-tile 0..3
  const int wn    = wid >> 2;   // N sub-tile 0..1
  const int m0    = blockIdx.y * 64;
  const int n0    = blockIdx.x * 32;
  const int lhalf = lane >> 4;
  const int l16   = lane & 15;

  const int arow = tid >> 2;        // 0..63
  const int acg  = (tid & 3) * 8;   // 0,8,16,24
  const int gm   = m0 + arow;
  const int bn   = n0 + wn * 16 + l16;
  const int mrow = wm * 16 + l16;

  v8f acc = {0.f, 0.f, 0.f, 0.f, 0.f, 0.f, 0.f, 0.f};

  if (!EDGE) {
    const _Float16* ap = A + (size_t)gm * lda + acg;
    // prologue: stage k0 = 0
    *(v8h*)(&As[0][arow * 32 + acg]) = *(const v8h*)ap;
    __syncthreads();
    int cur = 0;
    for (int k0 = 0; k0 < K; k0 += 32) {
      const bool has_next = (k0 + 32) < K;   // uniform
      v8h nv;
      if (has_next) {
        nv = *(const v8h*)(ap + k0 + 32);
        if (k0 + 64 < K) __builtin_prefetch(ap + k0 + 64, 0, 1);
      }
      v16h af = load_afrag(&As[cur][0], mrow, lhalf);
      v16h bf = load_bfrag_fast(W, bn, K, k0 + lhalf * 16);
      acc = wmma_f16f32(af, bf, acc);
      if (has_next) *(v8h*)(&As[cur ^ 1][arow * 32 + acg]) = nv;
      __syncthreads();
      cur ^= 1;
    }
  } else {
    for (int k0 = 0; k0 < K; k0 += 32) {
      v8h av;
      if (gm < M && k0 + acg + 8 <= K) {
        av = *(const v8h*)(A + (size_t)gm * lda + k0 + acg);
      } else {
        #pragma unroll
        for (int j = 0; j < 8; ++j) {
          int gk = k0 + acg + j;
          int cm = gm < M ? gm : 0;
          int ck = gk < K ? gk : 0;
          _Float16 v = A[(size_t)cm * lda + ck];
          av[j] = (gm < M && gk < K) ? v : (_Float16)0;
        }
      }
      *(v8h*)(&As[0][arow * 32 + acg]) = av;
      __syncthreads();
      v16h af = load_afrag(&As[0][0], mrow, lhalf);
      v16h bf = load_bfrag_edge(W, bn, N, K, k0 + lhalf * 16);
      acc = wmma_f16f32(af, bf, acc);
      __syncthreads();
    }
  }

  // Epilogue (D layout: vgpr r -> m = r + 8*half, n = lane&15)
  #pragma unroll
  for (int r = 0; r < 8; ++r) {
    int m = m0 + wm * 16 + r + lhalf * 8;
    int n = bn;
    if (EDGE && (m >= M || n >= N)) continue;
    float v = acc[r];
    if (bias) v += bias[n];
    if (res)  v += res[(size_t)m * ldr + n];
    v = act_apply(v, act);
    if (Cf) Cf[(size_t)m * ldc + n] = v;
    if (Ch) Ch[(size_t)m * ldc + n] = (_Float16)v;
  }
}

// ---------------------------------------------------------------------------
// Implicit-GEMM conv, kernel 4x4, stride 2, pad 1.
// grid.z = image; M = OH*OW, N = OC, K = IC*16. OW is a power of two.
// Spatial gather: clamped unconditional loads + select (no exec branching).
// EDGE=false requires K%32==0 and N%32==0 (conv1, the dominant kernel).
// ---------------------------------------------------------------------------
template <bool EDGE>
__global__ void __launch_bounds__(256)
jepa_conv_s2p1k4(const void* __restrict__ src, int src_is_video,
                 const _Float16* __restrict__ W, const float* __restrict__ bias,
                 float* __restrict__ outF, _Float16* __restrict__ outH,
                 int IC, int IH, int IW, int OC, int OH, int OW, int owshift,
                 int act) {
  __shared__ __align__(16) _Float16 As[2][64 * 32];

  const int tid   = threadIdx.x;
  const int lane  = tid & 31;
  const int wid   = tid >> 5;
  const int wm    = wid & 3;
  const int wn    = wid >> 2;
  const int bt    = blockIdx.z;
  const int m0    = blockIdx.y * 64;
  const int n0    = blockIdx.x * 32;
  const int lhalf = lane >> 4;
  const int l16   = lane & 15;

  const int M = OH * OW;
  const int K = IC * 16;
  const int N = OC;

  const float*    srcF = (const float*)src;
  const _Float16* srcH = (const _Float16*)src;

  const int arow = tid >> 2;
  const int acg  = (tid & 3) * 8;
  const int p    = m0 + arow;
  const int oy   = p >> owshift;
  const int ox   = p & (OW - 1);
  const int bn   = n0 + wn * 16 + l16;
  const int mrow = wm * 16 + l16;

  // Branch-free im2col gather of 8 consecutive k values for pixel p
  auto gather = [&](int kbase) -> v8h {
    v8h av;
    #pragma unroll
    for (int j = 0; j < 8; ++j) {
      int gk = kbase + j;
      int ic = gk >> 4;
      int r4 = gk & 15;
      int kh = r4 >> 2, kw = r4 & 3;
      int iy = oy * 2 - 1 + kh;
      int ix = ox * 2 - 1 + kw;
      bool inb = (iy >= 0) && (iy < IH) && (ix >= 0) && (ix < IW) &&
                 (EDGE ? ((p < M) && (gk < K)) : true);
      int ciy = min(max(iy, 0), IH - 1);
      int cix = min(max(ix, 0), IW - 1);
      int cic = EDGE ? (gk < K ? ic : 0) : ic;
      size_t sidx = (((size_t)bt * IC + cic) * IH + ciy) * IW + cix;
      float v = src_is_video ? (2.0f * srcF[sidx] - 1.0f) : (float)srcH[sidx];
      av[j] = inb ? (_Float16)v : (_Float16)0;
    }
    return av;
  };

  v8f acc = {0.f, 0.f, 0.f, 0.f, 0.f, 0.f, 0.f, 0.f};

  if (!EDGE) {
    // prologue: stage k0 = 0; then pipeline with ping-pong buffers
    *(v8h*)(&As[0][arow * 32 + acg]) = gather(acg);
    __syncthreads();
    int cur = 0;
    for (int k0 = 0; k0 < K; k0 += 32) {
      const bool has_next = (k0 + 32) < K;   // uniform
      v8h nv;
      if (has_next) nv = gather(k0 + 32 + acg);
      v16h af = load_afrag(&As[cur][0], mrow, lhalf);
      v16h bf = load_bfrag_fast(W, bn, K, k0 + lhalf * 16);
      acc = wmma_f16f32(af, bf, acc);
      if (has_next) *(v8h*)(&As[cur ^ 1][arow * 32 + acg]) = nv;
      __syncthreads();
      cur ^= 1;
    }
  } else {
    for (int k0 = 0; k0 < K; k0 += 32) {
      *(v8h*)(&As[0][arow * 32 + acg]) = gather(k0 + acg);
      __syncthreads();
      v16h af = load_afrag(&As[0][0], mrow, lhalf);
      v16h bf = load_bfrag_edge(W, bn, N, K, k0 + lhalf * 16);
      acc = wmma_f16f32(af, bf, acc);
      __syncthreads();
    }
  }

  #pragma unroll
  for (int r = 0; r < 8; ++r) {
    int pp = m0 + wm * 16 + r + lhalf * 8;
    int n  = bn;
    if (EDGE && (pp >= M || n >= N)) continue;
    int oyy = pp >> owshift, oxx = pp & (OW - 1);
    float v = acc[r] + bias[n];
    v = act_apply(v, act);
    size_t oidx = (((size_t)bt * OC + n) * OH + oyy) * OW + oxx;
    if (outF) outF[oidx] = v;
    if (outH) outH[oidx] = (_Float16)v;
  }
}

// ---------------------------------------------------------------------------
// LayerNorm (one wave per row) + optional activation; f32 in, f16 out.
// ---------------------------------------------------------------------------
__global__ void __launch_bounds__(256)
jepa_layernorm(const float* __restrict__ X, const float* __restrict__ g,
               const float* __restrict__ b, _Float16* __restrict__ Y,
               int M, int D, int act) {
  int row  = blockIdx.x * 8 + (threadIdx.x >> 5);
  int lane = threadIdx.x & 31;
  if (row >= M) return;
  const float* x = X + (size_t)row * D;
  float s = 0.f, s2 = 0.f;
  for (int j = lane; j < D; j += 32) {
    float v = x[j];
    s += v; s2 += v * v;
  }
  #pragma unroll
  for (int o = 16; o > 0; o >>= 1) {
    s  += __shfl_xor(s,  o, 32);
    s2 += __shfl_xor(s2, o, 32);
  }
  float inv = 1.0f / (float)D;
  float mu  = s * inv;
  float var = s2 * inv - mu * mu;
  float rst = rsqrtf(var + 1e-5f);
  _Float16* y = Y + (size_t)row * D;
  for (int j = lane; j < D; j += 32) {
    float v = (x[j] - mu) * rst * g[j] + b[j];
    y[j] = (_Float16)act_apply(v, act);
  }
}

// ---------------------------------------------------------------------------
// Attention: one wave per (batch, head). n=16 tokens, d=64.
// S = (Q K^T) * 0.125 via 2 WMMAs; causal softmax; O = P V via 4 WMMAs.
// V is transposed into LDS at stage time so P@V B-fragments are contiguous.
// ---------------------------------------------------------------------------
__global__ void __launch_bounds__(32)
jepa_attention(const _Float16* __restrict__ qkv, _Float16* __restrict__ out) {
  __shared__ __align__(16) _Float16 Qs[16 * 64];
  __shared__ __align__(16) _Float16 Ks[16 * 64];
  __shared__ __align__(16) _Float16 Vt[64 * 16];   // [dim][key]
  __shared__ float                  Sb[16 * 16];
  __shared__ __align__(16) _Float16 Pb[16 * 16];

  const int h    = blockIdx.x;   // head 0..7
  const int b    = blockIdx.y;   // batch 0..7
  const int lane = threadIdx.x;

  for (int cidx = lane; cidx < 128; cidx += 32) {
    int r  = cidx >> 3;         // token 0..15
    int c8 = (cidx & 7) * 8;    // dim chunk
    size_t base = ((size_t)(b * 16 + r)) * 1536 + h * 64 + c8;
    v8h q = *(const v8h*)(qkv + base);
    v8h k = *(const v8h*)(qkv + base + 512);
    v8h v = *(const v8h*)(qkv + base + 1024);
    *(v8h*)(Qs + r * 64 + c8) = q;
    *(v8h*)(Ks + r * 64 + c8) = k;
    #pragma unroll
    for (int j = 0; j < 8; ++j) Vt[(c8 + j) * 16 + r] = v[j];
  }
  __syncthreads();

  const int lhalf = lane >> 4;
  const int l16   = lane & 15;
  const int qrow  = l16 * 64;

  // S = Q @ K^T  (K-dim = 64)
  v8f sc = {0.f, 0.f, 0.f, 0.f, 0.f, 0.f, 0.f, 0.f};
  #pragma unroll
  for (int ks = 0; ks < 64; ks += 32) {
    v16h af = cat8(*(const v8h*)(Qs + qrow + ks + lhalf * 16),
                   *(const v8h*)(Qs + qrow + ks + lhalf * 16 + 8));
    v16h bf = cat8(*(const v8h*)(Ks + qrow + ks + lhalf * 16),
                   *(const v8h*)(Ks + qrow + ks + lhalf * 16 + 8));
    sc = wmma_f16f32(af, bf, sc);
  }
  #pragma unroll
  for (int r = 0; r < 8; ++r)
    Sb[(r + lhalf * 8) * 16 + l16] = sc[r] * 0.125f;
  __syncthreads();

  // Causal softmax, one row per lane (lanes 0..15)
  if (lane < 16) {
    int ri = lane;
    float mx = -1e30f;
    for (int j = 0; j <= ri; ++j) mx = fmaxf(mx, Sb[ri * 16 + j]);
    float e[16];
    float sum = 0.f;
    #pragma unroll
    for (int j = 0; j < 16; ++j) {
      float v = (j <= ri) ? __expf(Sb[ri * 16 + j] - mx) : 0.f;
      e[j] = v; sum += v;
    }
    float isum = 1.0f / sum;
    #pragma unroll
    for (int j = 0; j < 16; ++j) Pb[ri * 16 + j] = (_Float16)(e[j] * isum);
  }
  __syncthreads();

  // O = P @ V  (K-dim = 16, zero-padded to 32)
  v16h af = (lhalf == 0) ? cat8(*(const v8h*)(Pb + l16 * 16), *(const v8h*)(Pb + l16 * 16 + 8))
                         : cat8(v8h_zero(), v8h_zero());
  #pragma unroll
  for (int t = 0; t < 4; ++t) {
    v8h bl = v8h_zero(), bh = v8h_zero();
    if (lhalf == 0) {
      bl = *(const v8h*)(Vt + (t * 16 + l16) * 16);
      bh = *(const v8h*)(Vt + (t * 16 + l16) * 16 + 8);
    }
    v16h bf = cat8(bl, bh);
    v8f oc = {0.f, 0.f, 0.f, 0.f, 0.f, 0.f, 0.f, 0.f};
    oc = wmma_f16f32(af, bf, oc);
    #pragma unroll
    for (int r = 0; r < 8; ++r) {
      int m = r + lhalf * 8;
      out[((size_t)(b * 16 + m)) * 512 + h * 64 + t * 16 + l16] = (_Float16)oc[r];
    }
  }
}

// ---------------------------------------------------------------------------
// moments 1x1 conv (first 4 channels) -> states_in (f16) and states_tg (f32)
// ---------------------------------------------------------------------------
__global__ void jepa_states(const float* __restrict__ h2, const float* __restrict__ qw,
                            const float* __restrict__ qb,
                            _Float16* __restrict__ sin16, float* __restrict__ stg) {
  int idx = blockIdx.x * blockDim.x + threadIdx.x;   // 136*4*1024
  if (idx >= 136 * 4 * 1024) return;
  int s  = idx & 1023;
  int c  = (idx >> 10) & 3;
  int bt = idx >> 12;
  const float* hb = h2 + (size_t)bt * 8 * 1024;
  float v = qb[c];
  #pragma unroll
  for (int ic = 0; ic < 8; ++ic) v += qw[c * 8 + ic] * hb[ic * 1024 + s];
  int bb = bt / 17, t = bt - bb * 17;
  if (t < 16) sin16[((size_t)(bb * 16 + t)) * 4096 + c * 1024 + s] = (_Float16)v;
  if (t >= 1) stg[((size_t)(bb * 16 + t - 1)) * 4096 + c * 1024 + s] = v;
}

__global__ void jepa_actions_f16(const float* __restrict__ act, _Float16* __restrict__ out) {
  int idx = blockIdx.x * blockDim.x + threadIdx.x;   // 128*8
  if (idx >= 128 * 8) return;
  int c = idx & 7;
  int row = idx >> 3;
  int b = row >> 4, t = row & 15;
  out[idx] = (_Float16)act[((size_t)(b * 17 + t)) * 8 + c];
}

__global__ void jepa_add_pos(float* __restrict__ x, const float* __restrict__ pos) {
  int idx = blockIdx.x * blockDim.x + threadIdx.x;   // 128*512
  if (idx >= 128 * 512) return;
  int n = idx & 511;
  int row = idx >> 9;
  int t = row & 15;
  x[idx] += pos[t * 512 + n];
}

// ---------------------------------------------------------------------------
// Smooth-L1 per-row mean, then masked mean -> scalar loss
// ---------------------------------------------------------------------------
__global__ void __launch_bounds__(256)
jepa_loss_rows(const float* __restrict__ pred, const float* __restrict__ tg,
               float* __restrict__ lrow) {
  int row = blockIdx.x;
  __shared__ float red[256];
  float s = 0.f;
  for (int j = threadIdx.x; j < 4096; j += 256) {
    float d  = pred[(size_t)row * 4096 + j] - tg[(size_t)row * 4096 + j];
    float ad = fabsf(d);
    s += (ad < 1.0f) ? 0.5f * ad * ad : (ad - 0.5f);
  }
  red[threadIdx.x] = s;
  __syncthreads();
  for (int o = 128; o > 0; o >>= 1) {
    if (threadIdx.x < o) red[threadIdx.x] += red[threadIdx.x + o];
    __syncthreads();
  }
  if (threadIdx.x == 0) lrow[row] = red[0] * (1.0f / 4096.0f);
}

__global__ void __launch_bounds__(128)
jepa_loss_final(const float* __restrict__ lrow, const unsigned char* __restrict__ masks,
                float* __restrict__ out) {
  int tid = threadIdx.x;            // 0..127 -> (b,t)
  int b = tid >> 4, t = tid & 15;
  float tm = (masks[b * 17 + t] && masks[b * 17 + t + 1]) ? 1.0f : 0.0f;
  float l  = lrow[tid];
  __shared__ float rs[128], rt[128], rl[128];
  rs[tid] = l * tm; rt[tid] = tm; rl[tid] = l;
  __syncthreads();
  for (int o = 64; o > 0; o >>= 1) {
    if (tid < o) { rs[tid] += rs[tid + o]; rt[tid] += rt[tid + o]; rl[tid] += rl[tid + o]; }
    __syncthreads();
  }
  if (tid == 0)
    out[0] = (rt[0] > 0.f) ? rs[0] / fmaxf(rt[0], 1.0f) : rl[0] * (1.0f / 128.0f);
}

// ---------------------------------------------------------------------------
// Host-side orchestration
// ---------------------------------------------------------------------------
extern "C" void kernel_launch(void* const* d_in, const int* in_sizes, int n_in,
                              void* d_out, int out_size, void* d_ws, size_t ws_size,
                              hipStream_t stream) {
  (void)in_sizes; (void)n_in; (void)out_size; (void)ws_size;

  // ---- inputs (setup_inputs dict order, params flattened in insertion order)
  const float* videos  = (const float*)d_in[0];
  const float* actions = (const float*)d_in[1];
  const unsigned char* masks = (const unsigned char*)d_in[2];
  const float* e0_w = (const float*)d_in[3];
  const float* e0_b = (const float*)d_in[4];
  const float* e1_w = (const float*)d_in[5];
  const float* e1_b = (const float*)d_in[6];
  const float* e2_w = (const float*)d_in[7];
  const float* e2_b = (const float*)d_in[8];
  const float* q_w  = (const float*)d_in[9];
  const float* q_b  = (const float*)d_in[10];
  const float* ae0_w = (const float*)d_in[11];
  const float* ae0_b = (const float*)d_in[12];
  const float* ae_ln_g = (const float*)d_in[13];
  const float* ae_ln_b = (const float*)d_in[14];
  const float* ae1_w = (const float*)d_in[15];
  const float* ae1_b = (const float*)d_in[16];
  const float* sp_w = (const float*)d_in[17];
  const float* sp_b = (const float*)d_in[18];
  const float* ap_w = (const float*)d_in[19];
  const float* ap_b = (const float*)d_in[20];
  const float* cp_w = (const float*)d_in[21];
  const float* cp_b = (const float*)d_in[22];
  const float* pos  = (const float*)d_in[23];
  const float *an_g[4], *an_b[4], *qkv_w[4], *ao_w[4], *ao_b[4],
              *fn_g[4], *fn_b[4], *f0_w[4], *f0_b[4], *f1_w[4], *f1_b[4];
  for (int i = 0; i < 4; ++i) {
    int p0 = 24 + 11 * i;
    an_g[i]  = (const float*)d_in[p0 + 0];
    an_b[i]  = (const float*)d_in[p0 + 1];
    qkv_w[i] = (const float*)d_in[p0 + 2];
    ao_w[i]  = (const float*)d_in[p0 + 3];
    ao_b[i]  = (const float*)d_in[p0 + 4];
    fn_g[i]  = (const float*)d_in[p0 + 5];
    fn_b[i]  = (const float*)d_in[p0 + 6];
    f0_w[i]  = (const float*)d_in[p0 + 7];
    f0_b[i]  = (const float*)d_in[p0 + 8];
    f1_w[i]  = (const float*)d_in[p0 + 9];
    f1_b[i]  = (const float*)d_in[p0 + 10];
  }
  const float* n_g  = (const float*)d_in[68];
  const float* n_b  = (const float*)d_in[69];
  const float* op_w = (const float*)d_in[70];
  const float* op_b = (const float*)d_in[71];
  float* out = (float*)d_out;

  // ---- workspace layout
  char* wsb = (char*)d_ws;
  size_t off = 0;
  auto alloc = [&](size_t bytes) -> void* {
    off = (off + 255) & ~(size_t)255;
    void* p = wsb + off;
    off += bytes;
    return p;
  };
  _Float16* h0 = (_Float16*)alloc((size_t)136 * 32 * 128 * 128 * 2);
  _Float16* h1 = (_Float16*)alloc((size_t)136 * 64 * 64 * 64 * 2);
  float*    h2 = (float*)   alloc((size_t)136 * 8 * 32 * 32 * 4);
  _Float16* sin16 = (_Float16*)alloc((size_t)128 * 4096 * 2);
  float*    stg   = (float*)   alloc((size_t)128 * 4096 * 4);
  _Float16* wt_e0 = (_Float16*)alloc((size_t)32 * 48 * 2);
  _Float16* wt_e1 = (_Float16*)alloc((size_t)64 * 512 * 2);
  _Float16* wt_e2 = (_Float16*)alloc((size_t)8 * 1024 * 2);
  _Float16* wt_ae0 = (_Float16*)alloc((size_t)256 * 8 * 2);
  _Float16* wt_ae1 = (_Float16*)alloc((size_t)256 * 256 * 2);
  _Float16* wt_sp = (_Float16*)alloc((size_t)512 * 4096 * 2);
  _Float16* wt_ap = (_Float16*)alloc((size_t)512 * 256 * 2);
  _Float16* wt_cp = (_Float16*)alloc((size_t)512 * 1024 * 2);
  _Float16* wt_op = (_Float16*)alloc((size_t)4096 * 512 * 2);
  _Float16 *wt_qkv[4], *wt_ao[4], *wt_f0[4], *wt_f1[4];
  for (int i = 0; i < 4; ++i) {
    wt_qkv[i] = (_Float16*)alloc((size_t)1536 * 512 * 2);
    wt_ao[i]  = (_Float16*)alloc((size_t)512 * 512 * 2);
    wt_f0[i]  = (_Float16*)alloc((size_t)2048 * 512 * 2);
    wt_f1[i]  = (_Float16*)alloc((size_t)512 * 2048 * 2);
  }
  _Float16* act16  = (_Float16*)alloc((size_t)128 * 8 * 2);
  float*    a0f    = (float*)   alloc((size_t)128 * 256 * 4);
  _Float16* a1h    = (_Float16*)alloc((size_t)128 * 256 * 2);
  _Float16* a2h    = (_Float16*)alloc((size_t)128 * 256 * 2);
  _Float16* concat = (_Float16*)alloc((size_t)128 * 1024 * 2);
  float*    xf     = (float*)   alloc((size_t)128 * 512 * 4);
  _Float16* xnh    = (_Float16*)alloc((size_t)128 * 512 * 2);
  _Float16* qkvh   = (_Float16*)alloc((size_t)128 * 1536 * 2);
  _Float16* attnh  = (_Float16*)alloc((size_t)128 * 512 * 2);
  _Float16* mlph   = (_Float16*)alloc((size_t)128 * 2048 * 2);
  float*    predf  = (float*)   alloc((size_t)128 * 4096 * 4);
  float*    lrow   = (float*)   alloc((size_t)128 * 4);

  // ---- weight conversion (straight f32 -> f16, layout preserved N x K)
  auto cvt = [&](const float* W, _Float16* dst, int total) {
    jepa_w_f16<<<(total / 4 + 255) / 256, 256, 0, stream>>>(W, dst, total);
  };
  cvt(e0_w, wt_e0, 32 * 48);
  cvt(e1_w, wt_e1, 64 * 512);
  cvt(e2_w, wt_e2, 8 * 1024);
  cvt(ae0_w, wt_ae0, 256 * 8);
  cvt(ae1_w, wt_ae1, 256 * 256);
  cvt(sp_w, wt_sp, 512 * 4096);
  cvt(ap_w, wt_ap, 512 * 256);
  cvt(cp_w, wt_cp, 512 * 1024);
  cvt(op_w, wt_op, 4096 * 512);
  for (int i = 0; i < 4; ++i) {
    cvt(qkv_w[i], wt_qkv[i], 1536 * 512);
    cvt(ao_w[i],  wt_ao[i],  512 * 512);
    cvt(f0_w[i],  wt_f0[i],  2048 * 512);
    cvt(f1_w[i],  wt_f1[i],  512 * 2048);
  }

  // ---- encoder convs (implicit GEMM, WMMA)
  // conv0: K=48 (not /32)  -> EDGE ; conv1: aligned -> fast ; conv2: N=8 -> EDGE
  jepa_conv_s2p1k4<true><<<dim3(1, 256, 136), 256, 0, stream>>>(
      videos, 1, wt_e0, e0_b, nullptr, h0, 3, 256, 256, 32, 128, 128, 7, 1);
  jepa_conv_s2p1k4<false><<<dim3(2, 64, 136), 256, 0, stream>>>(
      h0, 0, wt_e1, e1_b, nullptr, h1, 32, 128, 128, 64, 64, 64, 6, 1);
  jepa_conv_s2p1k4<true><<<dim3(1, 16, 136), 256, 0, stream>>>(
      h1, 0, wt_e2, e2_b, h2, nullptr, 64, 64, 64, 8, 32, 32, 5, 0);

  // ---- states (1x1 moments conv, first 4 channels), actions
  jepa_states<<<(136 * 4 * 1024 + 255) / 256, 256, 0, stream>>>(h2, q_w, q_b, sin16, stg);
  jepa_actions_f16<<<(128 * 8 + 255) / 256, 256, 0, stream>>>(actions, act16);

  auto gemm = [&](const _Float16* A, int lda, const _Float16* W, const float* bias,
                  const float* res, int ldr, float* Cf, _Float16* Ch, int ldc,
                  int M, int N, int K, int act) {
    dim3 grid((N + 31) / 32, (M + 63) / 64, 1);
    bool aligned = (M % 64 == 0) && (N % 32 == 0) && (K % 32 == 0) && (lda % 8 == 0);
    if (aligned)
      jepa_gemm_f16<false><<<grid, 256, 0, stream>>>(A, lda, W, bias, res, ldr,
                                                     Cf, Ch, ldc, M, N, K, act);
    else
      jepa_gemm_f16<true><<<grid, 256, 0, stream>>>(A, lda, W, bias, res, ldr,
                                                    Cf, Ch, ldc, M, N, K, act);
  };

  // ---- action embedding MLP
  gemm(act16, 8, wt_ae0, ae0_b, nullptr, 0, a0f, nullptr, 256, 128, 256, 8, 0);
  jepa_layernorm<<<16, 256, 0, stream>>>(a0f, ae_ln_g, ae_ln_b, a1h, 128, 256, 2);
  gemm(a1h, 256, wt_ae1, ae1_b, nullptr, 0, nullptr, a2h, 256, 128, 256, 256, 0);

  // ---- projections into concat buffer, then combine + pos
  gemm(sin16, 4096, wt_sp, sp_b, nullptr, 0, nullptr, concat, 1024, 128, 512, 4096, 0);
  gemm(a2h, 256, wt_ap, ap_b, nullptr, 0, nullptr, concat + 512, 1024, 128, 512, 256, 0);
  gemm(concat, 1024, wt_cp, cp_b, nullptr, 0, xf, nullptr, 512, 128, 512, 1024, 0);
  jepa_add_pos<<<(128 * 512 + 255) / 256, 256, 0, stream>>>(xf, pos);

  // ---- transformer layers
  for (int i = 0; i < 4; ++i) {
    jepa_layernorm<<<16, 256, 0, stream>>>(xf, an_g[i], an_b[i], xnh, 128, 512, 0);
    gemm(xnh, 512, wt_qkv[i], nullptr, nullptr, 0, nullptr, qkvh, 1536, 128, 1536, 512, 0);
    jepa_attention<<<dim3(8, 8), 32, 0, stream>>>(qkvh, attnh);
    gemm(attnh, 512, wt_ao[i], ao_b[i], xf, 512, xf, nullptr, 512, 128, 512, 512, 0);
    jepa_layernorm<<<16, 256, 0, stream>>>(xf, fn_g[i], fn_b[i], xnh, 128, 512, 0);
    gemm(xnh, 512, wt_f0[i], f0_b[i], nullptr, 0, nullptr, mlph, 2048, 128, 2048, 512, 2);
    gemm(mlph, 2048, wt_f1[i], f1_b[i], xf, 512, xf, nullptr, 512, 128, 512, 2048, 0);
  }

  // ---- head + loss
  jepa_layernorm<<<16, 256, 0, stream>>>(xf, n_g, n_b, xnh, 128, 512, 0);
  gemm(xnh, 512, wt_op, op_b, nullptr, 0, predf, nullptr, 4096, 128, 4096, 512, 0);
  jepa_loss_rows<<<128, 256, 0, stream>>>(predf, stg, lrow);
  jepa_loss_final<<<1, 128, 0, stream>>>(lrow, masks, out);
}
